// CEAlignmentInformation_14224931684632
// MI455X (gfx1250) — compile-verified, matrix operand order checked by reading, not run on hardware
//
#include <hip/hip_runtime.h>

// ---------------------------------------------------------------------------
// Problem sizes (fixed by the reference)
// ---------------------------------------------------------------------------
#define BSZ   1024
#define CL    10
#define EDIM  64
#define HDIM  1024
#define XDIM  768
#define ECDIM 640                    // E*C
#define NBB   (1024 * 1024)          // B*B
#define POUT_ELEMS ((size_t)NBB * CL)

typedef __attribute__((ext_vector_type(16))) __bf16 v16bf;
typedef __attribute__((ext_vector_type(8)))  float  v8f;

// round-to-nearest-even f32 -> bf16 (bit trick, no __bf16 arithmetic needed)
__device__ __forceinline__ unsigned short f2bf(float f) {
  unsigned u = __float_as_uint(f);
  unsigned r = u + 0x7FFFu + ((u >> 16) & 1u);
  return (unsigned short)(r >> 16);
}
__device__ __forceinline__ unsigned pk2(float lo, float hi) {
  return (unsigned)f2bf(lo) | ((unsigned)f2bf(hi) << 16);
}

// ---------------------------------------------------------------------------
// Tiled bf16 WMMA GEMM:  out[M,N] = act(A[M,K] @ W[K,N] + bias[N])
// Block tile 64x64, K-step 32, 256 threads = 8 waves; each wave produces a
// 16x32 sub-tile with two v_wmma_f32_16x16x32_bf16.
// LDS tiles are stored in WMMA fragment order so each lane's fragment is one
// contiguous, 32B-aligned 16-element bf16 vector.
//   ISA 16-bit A 16x32 layout: lane = (m&15) + 16*laneHi,
//   laneHi = (k>>3)&1,  element e = ((k>>4)<<3) | (k&7)
// ---------------------------------------------------------------------------
template <int RELU>
__global__ __launch_bounds__(256)
void gemm_bf16_wmma(const float* __restrict__ A, const float* __restrict__ W,
                    const float* __restrict__ bias, float* __restrict__ out,
                    int M, int N, int K)
{
  __shared__ __align__(32) unsigned short ldsA[4][32][16]; // [msub][lane][e]
  __shared__ __align__(32) unsigned short ldsB[4][32][16]; // [nsub][lane][e]

  const int tid   = threadIdx.x;
  const int lane  = tid & 31;
  const int wave  = tid >> 5;
  const int mSub  = wave & 3;   // which 16-row sub-tile
  const int nPair = wave >> 2;  // which pair of 16-col sub-tiles
  const int rowBase = blockIdx.y * 64;
  const int colBase = blockIdx.x * 64;

  v8f acc0 = {};
  v8f acc1 = {};

  // A staging: thread -> row am (0..63), 8 consecutive k at ak
  const int am = tid >> 2;
  const int ak = (tid & 3) * 8;
  const int laneHiA = (ak >> 3) & 1;
  const int e0A     = (ak >> 4) << 3;
  // B staging: thread -> k row bk (0..31), 8 consecutive n at bn
  const int bk = tid >> 3;
  const int bn = (tid & 7) * 8;
  const int laneHiB = (bk >> 3) & 1;
  const int eB      = ((bk >> 4) << 3) | (bk & 7);

  const float* aptr = A + (size_t)(rowBase + am) * K + ak;
  const float* bptr = W + (size_t)bk * N + colBase + bn;

  for (int k0 = 0; k0 < K; k0 += 32) {
    float4 a0 = *(const float4*)(aptr);
    float4 a1 = *(const float4*)(aptr + 4);
    float4 b0 = *(const float4*)(bptr);
    float4 b1 = *(const float4*)(bptr + 4);
    if (k0 + 32 < K) {                      // global_prefetch_b8 next tiles
      __builtin_prefetch(aptr + 32, 0, 3);
      __builtin_prefetch(bptr + (size_t)32 * N, 0, 3);
    }
    aptr += 32;
    bptr += (size_t)32 * N;

    // A fragment store: 8 bf16 contiguous -> one 16B LDS store
    {
      uint4 pk = { pk2(a0.x, a0.y), pk2(a0.z, a0.w),
                   pk2(a1.x, a1.y), pk2(a1.z, a1.w) };
      *(uint4*)&ldsA[am >> 4][(am & 15) + (laneHiA << 4)][e0A] = pk;
    }
    // B fragment store: scattered across lanes (one b16 each)
    {
      float bv[8] = {b0.x, b0.y, b0.z, b0.w, b1.x, b1.y, b1.z, b1.w};
#pragma unroll
      for (int j = 0; j < 8; ++j) {
        int n = bn + j;
        ldsB[n >> 4][(n & 15) + (laneHiB << 4)][eB] = f2bf(bv[j]);
      }
    }
    __syncthreads();

    v16bf af  = *(const v16bf*)&ldsA[mSub][lane][0];
    v16bf bf0 = *(const v16bf*)&ldsB[nPair * 2 + 0][lane][0];
    v16bf bf1 = *(const v16bf*)&ldsB[nPair * 2 + 1][lane][0];
    acc0 = __builtin_amdgcn_wmma_f32_16x16x32_bf16(false, af, false, bf0,
                                                   (short)0, acc0, false, false);
    acc1 = __builtin_amdgcn_wmma_f32_16x16x32_bf16(false, af, false, bf1,
                                                   (short)0, acc1, false, false);
    __syncthreads();
  }

  // Epilogue. C layout: VGPR r -> row r + 8*(lane>>4); col = lane&15.
  const int laneHi = lane >> 4;
  const int n0 = colBase + (nPair * 2 + 0) * 16 + (lane & 15);
  const int n1 = colBase + (nPair * 2 + 1) * 16 + (lane & 15);
  const float bias0 = bias[n0];
  const float bias1 = bias[n1];
  const int m = rowBase + mSub * 16 + 8 * laneHi;
#pragma unroll
  for (int r = 0; r < 8; ++r) {
    float v0 = acc0[r] + bias0;
    float v1 = acc1[r] + bias1;
    if (RELU) { v0 = fmaxf(v0, 0.f); v1 = fmaxf(v1, 0.f); }
    out[(size_t)(m + r) * N + n0] = v0;
    out[(size_t)(m + r) * N + n1] = v1;
  }
}

// ---------------------------------------------------------------------------
// z-score over E=64 (ddof=1), f32 [B][C][E] -> bf16 [C][B][E]
// 4 rows per 256-thread block; 64 threads per row.
// ---------------------------------------------------------------------------
__global__ __launch_bounds__(256)
void zscore_kernel(const float* __restrict__ q, unsigned short* __restrict__ qn)
{
  __shared__ float red[256];
  const int tid = threadIdx.x;
  const int grp = tid >> 6;
  const int e   = tid & 63;
  const int rowIdx = blockIdx.x * 4 + grp;      // b*C + c
  const int b = rowIdx / CL;
  const int c = rowIdx % CL;

  const float v = q[(size_t)rowIdx * EDIM + e];
  red[tid] = v;
  __syncthreads();
#pragma unroll
  for (int s = 32; s > 0; s >>= 1) {
    if (e < s) red[tid] += red[tid + s];
    __syncthreads();
  }
  const float mu = red[grp * 64] * (1.f / 64.f);
  __syncthreads();
  const float d = v - mu;
  red[tid] = d * d;
  __syncthreads();
#pragma unroll
  for (int s = 32; s > 0; s >>= 1) {
    if (e < s) red[tid] += red[tid + s];
    __syncthreads();
  }
  const float var = red[grp * 64] * (1.f / 63.f);   // unbiased
  const float z = d * rsqrtf(var + 1e-8f);
  qn[((size_t)c * BSZ + b) * EDIM + e] = f2bf(z);
}

// ---------------------------------------------------------------------------
// Affinity: aff[c][m][n] = (1/8) * sum_e q1n[c][m][e] * q2n[c][n][e]
// Batched bf16 WMMA GEMM, K=64 staged entirely in LDS (2 k-steps).
// ---------------------------------------------------------------------------
__global__ __launch_bounds__(256)
void affinity_wmma(const unsigned short* __restrict__ q1n,
                   const unsigned short* __restrict__ q2n,
                   float* __restrict__ aff, float scale)
{
  __shared__ __align__(32) unsigned short ldsA[2][4][32][16]; // [ksub][msub][lane][e]
  __shared__ __align__(32) unsigned short ldsB[2][4][32][16]; // [ksub][nsub][lane][e]

  const int tid   = threadIdx.x;
  const int lane  = tid & 31;
  const int wave  = tid >> 5;
  const int mSub  = wave & 3;
  const int nPair = wave >> 2;
  const int c       = blockIdx.z;
  const int rowBase = blockIdx.y * 64;
  const int colBase = blockIdx.x * 64;

  // Staging: thread -> local row rr (0..63), 16 consecutive k at kb.
  // k-group of 16: first 8 -> laneHi 0, last 8 -> laneHi 1; e0 = ((kb>>4)&1)*8
  const int rr   = tid >> 2;
  const int kb   = (tid & 3) * 16;
  const int ksub = kb >> 5;
  const int e0   = ((kb >> 4) & 1) * 8;

  {
    const uint4* ap = (const uint4*)(q1n + ((size_t)c * BSZ + rowBase + rr) * EDIM + kb);
    const uint4* bp = (const uint4*)(q2n + ((size_t)c * BSZ + colBase + rr) * EDIM + kb);
    uint4 a0 = ap[0], a1 = ap[1];
    uint4 b0 = bp[0], b1 = bp[1];
    *(uint4*)&ldsA[ksub][rr >> 4][(rr & 15)][e0]      = a0;
    *(uint4*)&ldsA[ksub][rr >> 4][(rr & 15) + 16][e0] = a1;
    *(uint4*)&ldsB[ksub][rr >> 4][(rr & 15)][e0]      = b0;
    *(uint4*)&ldsB[ksub][rr >> 4][(rr & 15) + 16][e0] = b1;
  }
  __syncthreads();

  v8f acc0 = {};
  v8f acc1 = {};
#pragma unroll
  for (int ks = 0; ks < 2; ++ks) {
    v16bf af  = *(const v16bf*)&ldsA[ks][mSub][lane][0];
    v16bf bf0 = *(const v16bf*)&ldsB[ks][nPair * 2 + 0][lane][0];
    v16bf bf1 = *(const v16bf*)&ldsB[ks][nPair * 2 + 1][lane][0];
    acc0 = __builtin_amdgcn_wmma_f32_16x16x32_bf16(false, af, false, bf0,
                                                   (short)0, acc0, false, false);
    acc1 = __builtin_amdgcn_wmma_f32_16x16x32_bf16(false, af, false, bf1,
                                                   (short)0, acc1, false, false);
  }

  const int laneHi = lane >> 4;
  const int n0 = colBase + (nPair * 2 + 0) * 16 + (lane & 15);
  const int n1 = colBase + (nPair * 2 + 1) * 16 + (lane & 15);
  const int m  = rowBase + mSub * 16 + 8 * laneHi;
  float* base = aff + (size_t)c * NBB;
#pragma unroll
  for (int r = 0; r < 8; ++r) {
    base[(size_t)(m + r) * BSZ + n0] = acc0[r] * scale;
    base[(size_t)(m + r) * BSZ + n1] = acc1[r] * scale;
  }
}

// ---------------------------------------------------------------------------
// Deterministic per-label max: partial (64 blocks/label) then final.
// ---------------------------------------------------------------------------
__global__ __launch_bounds__(256)
void max_partial(const float* __restrict__ aff, float* __restrict__ pmax)
{
  __shared__ float red[256];
  const int c = blockIdx.y;
  const float* p = aff + (size_t)c * NBB;
  float m = -3.4e38f;
  for (int i = blockIdx.x * 256 + threadIdx.x; i < NBB; i += 64 * 256)
    m = fmaxf(m, p[i]);
  red[threadIdx.x] = m;
  __syncthreads();
#pragma unroll
  for (int s = 128; s > 0; s >>= 1) {
    if (threadIdx.x < s) red[threadIdx.x] = fmaxf(red[threadIdx.x], red[threadIdx.x + s]);
    __syncthreads();
  }
  if (threadIdx.x == 0) pmax[c * 64 + blockIdx.x] = red[0];
}

__global__ void max_final(const float* __restrict__ pmax, float* __restrict__ maxV)
{
  __shared__ float red[64];
  const int c = blockIdx.x, t = threadIdx.x;
  red[t] = pmax[c * 64 + t];
  __syncthreads();
#pragma unroll
  for (int s = 32; s > 0; s >>= 1) {
    if (t < s) red[t] = fmaxf(red[t], red[t + s]);
    __syncthreads();
  }
  if (t == 0) maxV[c] = red[0];
}

// ---------------------------------------------------------------------------
// A = exp(aff - max[c]); write A to d_out ([b][d][c], 10 contiguous floats per
// lane) and to P scratch ([c][b][d]); per-block partial sums for sum(A).
// Block = (d-chunk of 256, b); blockId = b*4 + chunk.
// ---------------------------------------------------------------------------
__global__ __launch_bounds__(256)
void exp_kernel(const float* __restrict__ aff, const float* __restrict__ maxV,
                float* __restrict__ P, float* __restrict__ Aout,
                float* __restrict__ psum)
{
  __shared__ float red[256];
  const int tid = threadIdx.x;
  const int b = blockIdx.y;
  const int d = blockIdx.x * 256 + tid;
  const int blkId = b * 4 + blockIdx.x;

  float av[CL];
#pragma unroll
  for (int c = 0; c < CL; ++c) {
    const size_t idx = ((size_t)c * BSZ + b) * BSZ + d;
    float a = __expf(aff[idx] - maxV[c]);
    P[idx] = a;
    av[c] = a;
  }
#pragma unroll
  for (int c = 0; c < CL; ++c)
    Aout[((size_t)b * BSZ + d) * CL + c] = av[c];

#pragma unroll
  for (int c = 0; c < CL; ++c) {
    __syncthreads();
    red[tid] = av[c];
    __syncthreads();
#pragma unroll
    for (int s = 128; s > 0; s >>= 1) {
      if (tid < s) red[tid] += red[tid + s];
      __syncthreads();
    }
    if (tid == 0) psum[c * 4096 + blkId] = red[0];
  }
}

__global__ __launch_bounds__(256)
void sum_final(const float* __restrict__ psum, float* __restrict__ sumA)
{
  __shared__ float red[256];
  const int c = blockIdx.x, tid = threadIdx.x;
  float s = 0.f;
  for (int i = tid; i < 4096; i += 256) s += psum[c * 4096 + i];
  red[tid] = s;
  __syncthreads();
#pragma unroll
  for (int st = 128; st > 0; st >>= 1) {
    if (tid < st) red[tid] += red[tid + st];
    __syncthreads();
  }
  if (tid == 0) sumA[c] = red[0];
}

__global__ __launch_bounds__(256)
void scale_kernel(float* __restrict__ P, const float* __restrict__ sumA)
{
  const int c = blockIdx.y;
  const size_t i = (size_t)blockIdx.x * 256 + threadIdx.x;
  P[(size_t)c * NBB + i] = P[(size_t)c * NBB + i] / sumA[c];
}

// ---------------------------------------------------------------------------
// Sinkhorn passes on P in [C][B][B] layout (fully coalesced both ways).
// ---------------------------------------------------------------------------
__global__ __launch_bounds__(256)
void sinkhorn_row(float* __restrict__ P, const float* __restrict__ p1)
{
  __shared__ float red[256];
  const int blk = blockIdx.x;        // c*B + b
  const int b = blk & (BSZ - 1);
  const int c = blk >> 10;
  float* row = P + ((size_t)c * BSZ + b) * BSZ;
  const int tid = threadIdx.x;
  float v0 = row[tid], v1 = row[tid + 256], v2 = row[tid + 512], v3 = row[tid + 768];
  red[tid] = (v0 + v1) + (v2 + v3);
  __syncthreads();
#pragma unroll
  for (int s = 128; s > 0; s >>= 1) {
    if (tid < s) red[tid] += red[tid + s];
    __syncthreads();
  }
  const float scale = p1[b * CL + c] / (red[0] + 1e-12f);
  row[tid]       = v0 * scale;
  row[tid + 256] = v1 * scale;
  row[tid + 512] = v2 * scale;
  row[tid + 768] = v3 * scale;
}

__global__ __launch_bounds__(256)
void sinkhorn_col(float* __restrict__ P, const float* __restrict__ p2)
{
  const int c = blockIdx.y;
  const int d = blockIdx.x * 256 + threadIdx.x;
  float* base = P + (size_t)c * NBB;
  float s = 0.f;
#pragma unroll 4
  for (int b = 0; b < BSZ; ++b) s += base[(size_t)b * BSZ + d];
  const float scale = p2[d * CL + c] / (s + 1e-12f);
#pragma unroll 4
  for (int b = 0; b < BSZ; ++b) base[(size_t)b * BSZ + d] *= scale;
}

// [C][B][B] -> reference [B][B][C]; each lane writes 10 contiguous floats.
__global__ __launch_bounds__(256)
void transpose_out(const float* __restrict__ P, float* __restrict__ Pout)
{
  const int b = blockIdx.y;
  const int d = blockIdx.x * 256 + threadIdx.x;
  float v[CL];
#pragma unroll
  for (int c = 0; c < CL; ++c) v[c] = P[((size_t)c * BSZ + b) * BSZ + d];
#pragma unroll
  for (int c = 0; c < CL; ++c) Pout[((size_t)b * BSZ + d) * CL + c] = v[c];
}

// ---------------------------------------------------------------------------
// Host-side orchestration
// ---------------------------------------------------------------------------
extern "C" void kernel_launch(void* const* d_in, const int* in_sizes, int n_in,
                              void* d_out, int out_size, void* d_ws, size_t ws_size,
                              hipStream_t stream)
{
  (void)in_sizes; (void)n_in; (void)out_size; (void)ws_size;

  const float* x1 = (const float*)d_in[0];
  const float* x2 = (const float*)d_in[1];
  const float* p1 = (const float*)d_in[2];   // [B][C]
  const float* p2 = (const float*)d_in[3];   // [B][C]
  const float* m1_W0 = (const float*)d_in[4];  const float* m1_b0 = (const float*)d_in[5];
  const float* m1_W1 = (const float*)d_in[6];  const float* m1_b1 = (const float*)d_in[7];
  const float* m1_W2 = (const float*)d_in[8];  const float* m1_b2 = (const float*)d_in[9];
  const float* m1_Wo = (const float*)d_in[10]; const float* m1_bo = (const float*)d_in[11];
  const float* m2_W0 = (const float*)d_in[12]; const float* m2_b0 = (const float*)d_in[13];
  const float* m2_W1 = (const float*)d_in[14]; const float* m2_b1 = (const float*)d_in[15];
  const float* m2_W2 = (const float*)d_in[16]; const float* m2_b2 = (const float*)d_in[17];
  const float* m2_Wo = (const float*)d_in[18]; const float* m2_bo = (const float*)d_in[19];

  // Workspace layout (dead buffers overlap later big tensors):
  //  [0,40MB)   aff [C][B][B] f32     (hA, hB live here during the MLP phase)
  //  [40,80MB)  Pws [C][B][B] f32     (q1, q2 live here until zscore is done)
  //  [80MB..)   q1n/q2n bf16, reduction scratch
  char* w = (char*)d_ws;
  float* aff = (float*)w;
  float* hA  = aff;
  float* hB  = aff + (size_t)(1 << 20);                 // +4MB (in floats)
  float* Pws = (float*)(w + (size_t)40 * 1024 * 1024);
  float* q1  = Pws;                                     // [B][E*C]
  float* q2  = Pws + (size_t)BSZ * ECDIM;
  unsigned short* q1n = (unsigned short*)(w + (size_t)80 * 1024 * 1024); // [C][B][E]
  unsigned short* q2n = q1n + (size_t)CL * BSZ * EDIM;
  float* pmax = (float*)(q2n + (size_t)CL * BSZ * EDIM);
  float* maxV = pmax + 640;
  float* psum = maxV + 16;
  float* sumA = psum + (size_t)CL * 4096;

  float* Pout = (float*)d_out;
  float* Aout = Pout + POUT_ELEMS;

  const dim3 blk(256);

  // MLP 1: x1 -> q1
  gemm_bf16_wmma<1><<<dim3(16, 16), blk, 0, stream>>>(x1, m1_W0, m1_b0, hA, BSZ, HDIM, XDIM);
  gemm_bf16_wmma<1><<<dim3(16, 16), blk, 0, stream>>>(hA, m1_W1, m1_b1, hB, BSZ, HDIM, HDIM);
  gemm_bf16_wmma<1><<<dim3(16, 16), blk, 0, stream>>>(hB, m1_W2, m1_b2, hA, BSZ, HDIM, HDIM);
  gemm_bf16_wmma<0><<<dim3(10, 16), blk, 0, stream>>>(hA, m1_Wo, m1_bo, q1, BSZ, ECDIM, HDIM);
  // MLP 2: x2 -> q2
  gemm_bf16_wmma<1><<<dim3(16, 16), blk, 0, stream>>>(x2, m2_W0, m2_b0, hA, BSZ, HDIM, XDIM);
  gemm_bf16_wmma<1><<<dim3(16, 16), blk, 0, stream>>>(hA, m2_W1, m2_b1, hB, BSZ, HDIM, HDIM);
  gemm_bf16_wmma<1><<<dim3(16, 16), blk, 0, stream>>>(hB, m2_W2, m2_b2, hA, BSZ, HDIM, HDIM);
  gemm_bf16_wmma<0><<<dim3(10, 16), blk, 0, stream>>>(hA, m2_Wo, m2_bo, q2, BSZ, ECDIM, HDIM);

  // z-score + bf16 pack into label-major layout
  zscore_kernel<<<(BSZ * CL) / 4, blk, 0, stream>>>(q1, q1n);
  zscore_kernel<<<(BSZ * CL) / 4, blk, 0, stream>>>(q2, q2n);

  // per-label affinity GEMMs (scale = 1/sqrt(64))
  affinity_wmma<<<dim3(16, 16, CL), blk, 0, stream>>>(q1n, q2n, aff, 0.125f);

  // per-label max, then A = exp(aff - max), P0 = A / sum(A)
  max_partial<<<dim3(64, CL), blk, 0, stream>>>(aff, pmax);
  max_final<<<CL, 64, 0, stream>>>(pmax, maxV);
  exp_kernel<<<dim3(4, BSZ), blk, 0, stream>>>(aff, maxV, Pws, Aout, psum);
  sum_final<<<CL, blk, 0, stream>>>(psum, sumA);
  scale_kernel<<<dim3(4096, CL), blk, 0, stream>>>(Pws, sumA);

  // 20 Sinkhorn iterations (row-normalize then col-normalize, per label)
  for (int it = 0; it < 20; ++it) {
    sinkhorn_row<<<CL * BSZ, blk, 0, stream>>>(Pws, p1);
    sinkhorn_col<<<dim3(4, CL), blk, 0, stream>>>(Pws, p2);
  }

  // emit P in the reference's [B][B][C] layout
  transpose_out<<<dim3(4, BSZ), blk, 0, stream>>>(Pws, Pout);
}
